// FusedTransformerLayer_55173149884695
// MI455X (gfx1250) — matrix operand order repaired and checked by reading, not compiled, not run on hardware
//
#include <hip/hip_runtime.h>
#include <cstddef>
#include <cstdint>

// ---------------------------------------------------------------------------
// Fused transformer layer for MI455X (gfx1250, wave32, WMMA bf16 16x16x32)
//   B=1, L=2048, D=1024, H=16, E=64, ROT=32, F_IN=7D=7168, F_OUT=5D=5120
// Async global->LDS DMA (ASYNCcnt) used for all LDS staging.
// ---------------------------------------------------------------------------

#define L_SEQ   2048
#define D_MODEL 1024
#define NH      16
#define HE      64
#define ROTD    32
#define F_IN    (7 * D_MODEL)   // 7168
#define F_OUT   (5 * D_MODEL)   // 5120
#define EPSF    1e-6f
#define NEGF    (-1e30f)

typedef __attribute__((ext_vector_type(16))) __bf16 v16bf;
typedef __attribute__((ext_vector_type(8)))  __bf16 v8bf;
typedef __attribute__((ext_vector_type(8)))  float  v8f;

__device__ __forceinline__ v16bf cat16(v8bf lo, v8bf hi) {
    return __builtin_shufflevector(lo, hi, 0,1,2,3,4,5,6,7,8,9,10,11,12,13,14,15);
}

__device__ __forceinline__ v8f wmma_bf16(v16bf a, v16bf b, v8f c) {
    // D = A(16x32 bf16) * B(32x16 bf16) + C(16x16 f32)
    return __builtin_amdgcn_wmma_f32_16x16x32_bf16(
        /*neg_a=*/false, a, /*neg_b=*/false, b,
        /*c_mod=*/(short)0, c, /*reuse_a=*/false, /*reuse_b=*/false);
}

// contiguous 16-bf16 load as two 16B loads (16B-aligned at all call sites)
__device__ __forceinline__ v16bf ld16(const __bf16* p) {
    v8bf lo = *(const v8bf*)p;
    v8bf hi = *(const v8bf*)(p + 8);
    return cat16(lo, hi);
}

// generic (flat) shared-memory pointer -> 32-bit LDS address for async DMA
__device__ __forceinline__ uint32_t lds_addr32(const void* p) {
    return (uint32_t)(uintptr_t)p;
}

// async DMA: 16 bytes global -> LDS, tracked by ASYNCcnt
__device__ __forceinline__ void async_g2l_b128(uint32_t lds_dst, const void* gsrc) {
    asm volatile("global_load_async_to_lds_b128 %0, %1, off"
                 :: "v"(lds_dst), "v"(gsrc) : "memory");
}
__device__ __forceinline__ void wait_asynccnt0() {
    asm volatile("s_wait_asynccnt 0x0" ::: "memory");
}

// ---------------------------------------------------------------------------
// f32 -> bf16 conversion (grid-stride)
// ---------------------------------------------------------------------------
__global__ void cvt_bf16_kernel(const float* __restrict__ in,
                                __bf16* __restrict__ out, int n) {
    int i = blockIdx.x * blockDim.x + threadIdx.x;
    int stride = gridDim.x * blockDim.x;
    for (; i < n; i += stride) out[i] = (__bf16)in[i];
}

// ---------------------------------------------------------------------------
// RMSNorm: xn = x * rsqrt(mean(x^2)+eps) * (1+scale), bf16 out. 1 block/row.
// ---------------------------------------------------------------------------
__global__ __launch_bounds__(256)
void rmsnorm_kernel(const float* __restrict__ x, const float* __restrict__ scale,
                    __bf16* __restrict__ xn) {
    const int l = blockIdx.x;
    const float* row = x + (size_t)l * D_MODEL;
    float s = 0.f;
    for (int d = threadIdx.x; d < D_MODEL; d += 256) { float v = row[d]; s += v * v; }
    __shared__ float red[256];
    red[threadIdx.x] = s;
    __syncthreads();
    for (int off = 128; off > 0; off >>= 1) {
        if (threadIdx.x < off) red[threadIdx.x] += red[threadIdx.x + off];
        __syncthreads();
    }
    const float rs = rsqrtf(red[0] * (1.0f / D_MODEL) + EPSF);
    __bf16* orow = xn + (size_t)l * D_MODEL;
    for (int d = threadIdx.x; d < D_MODEL; d += 256)
        orow[d] = (__bf16)(row[d] * rs * (1.0f + scale[d]));
}

// ---------------------------------------------------------------------------
// GEMM: C[MxN](f32) = A[MxK](bf16) * B[NxK](bf16)^T  (+ skip)
// Block 256 = 8 waves; block tile 128(M) x 64(N); wave tile 16(M) x 64(N).
// B K-slab DMA'd to LDS via async global->LDS, double-buffered so the DMA
// for tile k+1 overlaps the WMMAs on tile k.
// ---------------------------------------------------------------------------
#define BSTRIDE 40  // LDS row stride in bf16 for the 64x32 B tile (pad vs 32)

__global__ __launch_bounds__(256)
void gemm_bf16_kernel(const __bf16* __restrict__ A, const __bf16* __restrict__ B,
                      float* __restrict__ C, const float* __restrict__ skip,
                      int M, int N, int K, int add_skip) {
    __shared__ __bf16 bt[2][64 * BSTRIDE];

    const int tid  = threadIdx.x;
    const int wave = tid >> 5;
    const int lane = tid & 31;
    const int hl   = lane >> 4;     // half-wave id
    const int ln   = lane & 15;
    const int m0   = blockIdx.x * 128 + wave * 16;
    const int n0   = blockIdx.y * 64;
    const int m    = m0 + ln;       // A row for this lane
    const int klo  = hl ? 8 : 0;    // unused; A loads use kb too
    const int kb   = hl ? 16 : 0;   // A and B K-offset: contiguous k = 16h + s

    v8f acc0 = {}, acc1 = {}, acc2 = {}, acc3 = {};

    const int srow = tid >> 2;      // staging: row 0..63
    const int schk = tid & 3;       // staging: 8-bf16 chunk 0..3
    const __bf16* bsrc_base = B + (size_t)(n0 + srow) * K + schk * 8;
    const uint32_t ldst0 = lds_addr32(&bt[0][srow * BSTRIDE + schk * 8]);
    const uint32_t ldst1 = lds_addr32(&bt[1][srow * BSTRIDE + schk * 8]);

    // prologue: DMA tile k0=0 into buffer 0
    async_g2l_b128(ldst0, bsrc_base);
    wait_asynccnt0();
    __syncthreads();

    for (int k0 = 0; k0 < K; k0 += 32) {
        const int cur = (k0 >> 5) & 1;
        // kick off DMA of the next K slab into the other buffer
        if (k0 + 32 < K)
            async_g2l_b128(cur ? ldst0 : ldst1, bsrc_base + k0 + 32);

        // A fragment: lane m, K = kb..kb+15
        const __bf16* arow = A + (size_t)m * K + k0;
        v16bf a = cat16(*(const v8bf*)(arow + kb), *(const v8bf*)(arow + kb + 8));
        __builtin_prefetch(arow + 128, 0, 1);

        // B fragments from LDS: lane col n0+t*16+ln, K = kb..kb+15
        const __bf16* btc = &bt[cur][0];
        v16bf b0 = cat16(*(const v8bf*)(&btc[(ln     ) * BSTRIDE + kb]),
                         *(const v8bf*)(&btc[(ln     ) * BSTRIDE + kb + 8]));
        v16bf b1 = cat16(*(const v8bf*)(&btc[(ln + 16) * BSTRIDE + kb]),
                         *(const v8bf*)(&btc[(ln + 16) * BSTRIDE + kb + 8]));
        v16bf b2 = cat16(*(const v8bf*)(&btc[(ln + 32) * BSTRIDE + kb]),
                         *(const v8bf*)(&btc[(ln + 32) * BSTRIDE + kb + 8]));
        v16bf b3 = cat16(*(const v8bf*)(&btc[(ln + 48) * BSTRIDE + kb]),
                         *(const v8bf*)(&btc[(ln + 48) * BSTRIDE + kb + 8]));

        acc0 = wmma_bf16(a, b0, acc0);
        acc1 = wmma_bf16(a, b1, acc1);
        acc2 = wmma_bf16(a, b2, acc2);
        acc3 = wmma_bf16(a, b3, acc3);

        // this wave's next-tile DMA must be complete before anyone reads it
        wait_asynccnt0();
        __syncthreads();
    }

    // C layout: vgpr r -> row m0 + r + hl*8, lane -> col ln within each N tile
#pragma unroll
    for (int r = 0; r < 8; ++r) {
        const int mr = m0 + r + hl * 8;
        float* crow = C + (size_t)mr * N + n0;
        if (add_skip) {
            const float* srw = skip + (size_t)mr * N + n0;
            crow[ln]      = acc0[r] + srw[ln];
            crow[16 + ln] = acc1[r] + srw[16 + ln];
            crow[32 + ln] = acc2[r] + srw[32 + ln];
            crow[48 + ln] = acc3[r] + srw[48 + ln];
        } else {
            crow[ln]      = acc0[r];
            crow[16 + ln] = acc1[r];
            crow[32 + ln] = acc2[r];
            crow[48 + ln] = acc3[r];
        }
    }
}

// ---------------------------------------------------------------------------
// QKV post-processing: per (l, h): L2-norm(q,k)*sqrt(attn_scale), RoPE,
// head-major bf16 outputs q/k/v [H][L][E].
// ---------------------------------------------------------------------------
__global__ __launch_bounds__(64)
void qkv_post_kernel(const float* __restrict__ proj, const float* __restrict__ theta,
                     const float* __restrict__ attn_scale,
                     __bf16* __restrict__ qh, __bf16* __restrict__ kh,
                     __bf16* __restrict__ vh) {
    const int l = blockIdx.x, hh = blockIdx.y, e = threadIdx.x;
    const float* pr = proj + (size_t)l * F_IN;
    const float qv = pr[hh * HE + e];
    const float kv = pr[D_MODEL + hh * HE + e];
    const float vv = pr[2 * D_MODEL + hh * HE + e];

    __shared__ float sq[HE], sk[HE];
    sq[e] = qv * qv; sk[e] = kv * kv;
    __syncthreads();
    float qs = 0.f, ks = 0.f;
#pragma unroll 8
    for (int i = 0; i < HE; ++i) { qs += sq[i]; ks += sk[i]; }
    const float sa = sqrtf(attn_scale[hh]);   // attn_scale row 0
    const float qn = qv * sa * rsqrtf(qs + EPSF);
    const float kn = kv * sa * rsqrtf(ks + EPSF);
    __syncthreads();
    sq[e] = qn; sk[e] = kn;
    __syncthreads();

    const int r = e & (ROTD - 1);
    const float th = theta[((size_t)hh * L_SEQ + l) * ROTD + r];
    const float c = cosf(th), sn = sinf(th);
    float qo, ko;
    if (e < ROTD) { qo = sq[e] * c - sq[e + ROTD] * sn; ko = sk[e] * c - sk[e + ROTD] * sn; }
    else          { qo = sq[e] * c + sq[e - ROTD] * sn; ko = sk[e] * c + sk[e - ROTD] * sn; }

    const size_t oi = ((size_t)hh * L_SEQ + l) * HE + e;
    qh[oi] = (__bf16)qo;
    kh[oi] = (__bf16)ko;
    vh[oi] = (__bf16)vv;
}

// ---------------------------------------------------------------------------
// GELU(up) -> bf16 into concat buffer h[:, D:5D]
// ---------------------------------------------------------------------------
__global__ void gelu_up_kernel(const float* __restrict__ proj,
                               __bf16* __restrict__ hbuf, int total) {
    int i = blockIdx.x * blockDim.x + threadIdx.x;
    const int stride = gridDim.x * blockDim.x;
    for (; i < total; i += stride) {
        const int l = i >> 12;          // / 4096
        const int j = i & 4095;
        const float v = proj[(size_t)l * F_IN + 3 * D_MODEL + j];
        const float g = 0.5f * v * (1.0f + tanhf(0.7978845608028654f *
                                                 (v + 0.044715f * v * v * v)));
        hbuf[(size_t)l * F_OUT + D_MODEL + j] = (__bf16)g;
    }
}

// ---------------------------------------------------------------------------
// Causal flash attention: one wave per (16-query tile, head).
// Scores: QK^T via WMMA (K=64 -> 2 steps); online softmax via wave shuffles;
// PV via WMMA with K=32 (two 16-key score tiles per iteration).
// P transposed C-layout -> A-layout through LDS; V staged in LDS via async DMA
// (instruction immediate offset advances global and LDS addresses together).
// ---------------------------------------------------------------------------
#define VSTR 68   // LDS row stride (bf16) for 32x64 V tile

__global__ __launch_bounds__(32)
void attn_kernel(const __bf16* __restrict__ qh, const __bf16* __restrict__ kh,
                 const __bf16* __restrict__ vh, __bf16* __restrict__ hbuf) {
    __shared__ __bf16 p_lds[16 * 32];
    __shared__ __bf16 vt[32 * VSTR];

    const int qt = blockIdx.x;       // 0..127
    const int hh = blockIdx.y;       // 0..15
    const int lane = threadIdx.x;
    const int hl = lane >> 4, ln = lane & 15;
    const int klo = hl ? 8 : 0;
    const int kb2 = hl ? 16 : 0;
    const int qbase = qt * 16;

    // Q A-fragments (held for whole tile): rows qbase+ln, E split 0..31 / 32..63
    const __bf16* qrow = qh + ((size_t)hh * L_SEQ + qbase + ln) * HE;
    const v16bf aq0 = cat16(*(const v8bf*)(qrow + kb2),      *(const v8bf*)(qrow + kb2 + 8));
    const v16bf aq1 = cat16(*(const v8bf*)(qrow + 32 + kb2), *(const v8bf*)(qrow + 40 + kb2));

    const uint32_t vt_lds = lds_addr32(&vt[lane * VSTR]);

    float rm[8], rl[8];
#pragma unroll
    for (int r = 0; r < 8; ++r) { rm[r] = NEGF; rl[r] = 0.f; }
    v8f o0 = {}, o1 = {}, o2 = {}, o3 = {};

    const int niter = (qt + 2) >> 1;          // ceil((qt+1)/2) 16-key tile pairs
    for (int it = 0; it < niter; ++it) {
        const int kb = it * 32;

        // ---- kick off async DMA of V[kb..kb+31][0..63] into LDS ----
        {
            const __bf16* vrow = vh + ((size_t)hh * L_SEQ + kb + lane) * HE;
            // offset immediate applies to BOTH global and LDS addresses; both
            // sides advance by the same 16B here (row-contiguous copy).
            asm volatile(
                "global_load_async_to_lds_b128 %0, %1, off\n\t"
                "global_load_async_to_lds_b128 %0, %1, off offset:16\n\t"
                "global_load_async_to_lds_b128 %0, %1, off offset:32\n\t"
                "global_load_async_to_lds_b128 %0, %1, off offset:48\n\t"
                "global_load_async_to_lds_b128 %0, %1, off offset:64\n\t"
                "global_load_async_to_lds_b128 %0, %1, off offset:80\n\t"
                "global_load_async_to_lds_b128 %0, %1, off offset:96\n\t"
                "global_load_async_to_lds_b128 %0, %1, off offset:112"
                :: "v"(vt_lds), "v"(vrow) : "memory");
        }

        // ---- scores for 32 keys (two 16-key tiles), overlapped with the DMA ----
        const __bf16* k0p = kh + ((size_t)hh * L_SEQ + kb + ln) * HE;
        const __bf16* k1p = k0p + 16 * HE;
        v8f s0 = {}, s1 = {};
        s0 = wmma_bf16(aq0, ld16(k0p + kb2), s0);
        s0 = wmma_bf16(aq1, ld16(k0p + 32 + kb2), s0);
        s1 = wmma_bf16(aq0, ld16(k1p + kb2), s1);
        s1 = wmma_bf16(aq1, ld16(k1p + 32 + kb2), s1);

        // ---- causal mask + online softmax ----
#pragma unroll
        for (int r = 0; r < 8; ++r) {
            const int qg = qbase + r + hl * 8;          // global query index
            float v0 = s0[r]; if (kb + ln > qg)      v0 = NEGF;
            float v1 = s1[r]; if (kb + 16 + ln > qg) v1 = NEGF;

            float t = fmaxf(v0, v1);
            t = fmaxf(t, __shfl_xor(t, 1, 32));
            t = fmaxf(t, __shfl_xor(t, 2, 32));
            t = fmaxf(t, __shfl_xor(t, 4, 32));
            t = fmaxf(t, __shfl_xor(t, 8, 32));
            const float mn = fmaxf(rm[r], t);           // always finite (diag in tile 0)
            const float sc = expf(rm[r] - mn);
            rl[r] *= sc;
            o0[r] = o0[r] * sc; o1[r] = o1[r] * sc;
            o2[r] = o2[r] * sc; o3[r] = o3[r] * sc;

            const float e0 = expf(v0 - mn);
            const float e1 = expf(v1 - mn);
            float rs = e0 + e1;
            rs += __shfl_xor(rs, 1, 32);
            rs += __shfl_xor(rs, 2, 32);
            rs += __shfl_xor(rs, 4, 32);
            rs += __shfl_xor(rs, 8, 32);
            rl[r] += rs;
            rm[r] = mn;

            const int mrow = r + hl * 8;                // P row within tile
            p_lds[mrow * 32 + ln]      = (__bf16)e0;
            p_lds[mrow * 32 + 16 + ln] = (__bf16)e1;
        }

        wait_asynccnt0();  // V tile landed in LDS
        __syncthreads();   // order LDS writes vs cross-lane reads

        // ---- P A-fragment (transpose from C layout through LDS) ----
        v16bf ap;
#pragma unroll
        for (int j = 0; j < 16; ++j) {
            const int kk = kb2 + j;
            ap[j] = p_lds[ln * 32 + kk];
        }

        // ---- V B-fragments + PV accumulation (K = 32 keys) ----
#pragma unroll
        for (int tN = 0; tN < 4; ++tN) {
            v16bf bv;
#pragma unroll
            for (int j = 0; j < 16; ++j)
                bv[j] = vt[(kb2 + j) * VSTR + tN * 16 + ln];
            if (tN == 0) o0 = wmma_bf16(ap, bv, o0);
            else if (tN == 1) o1 = wmma_bf16(ap, bv, o1);
            else if (tN == 2) o2 = wmma_bf16(ap, bv, o2);
            else o3 = wmma_bf16(ap, bv, o3);
        }
        __syncthreads();
    }

    // ---- normalize and write attn_out into concat buffer h[:, 0:D] ----
#pragma unroll
    for (int r = 0; r < 8; ++r) {
        const int qg = qbase + r + hl * 8;
        const float inv = 1.0f / rl[r];
        __bf16* dst = hbuf + (size_t)qg * F_OUT + hh * HE;
        dst[ln]      = (__bf16)(o0[r] * inv);
        dst[16 + ln] = (__bf16)(o1[r] * inv);
        dst[32 + ln] = (__bf16)(o2[r] * inv);
        dst[48 + ln] = (__bf16)(o3[r] * inv);
    }
}

// ---------------------------------------------------------------------------
// Host-side launch sequence
// ---------------------------------------------------------------------------
extern "C" void kernel_launch(void* const* d_in, const int* in_sizes, int n_in,
                              void* d_out, int out_size, void* d_ws, size_t ws_size,
                              hipStream_t stream) {
    const float* x          = (const float*)d_in[0];
    const float* theta      = (const float*)d_in[1];
    const float* scale      = (const float*)d_in[2];
    const float* w_in       = (const float*)d_in[3];
    const float* w_out      = (const float*)d_in[4];
    const float* attn_scale = (const float*)d_in[5];
    float* out = (float*)d_out;
    (void)in_sizes; (void)n_in; (void)out_size; (void)ws_size;

    char* ws = (char*)d_ws;
    size_t off = 0;
    auto alloc = [&](size_t bytes) -> void* {
        void* p = ws + off;
        off += (bytes + 255) & ~(size_t)255;
        return p;
    };
    __bf16* w_in_h  = (__bf16*)alloc((size_t)F_IN  * D_MODEL * 2);
    __bf16* w_out_h = (__bf16*)alloc((size_t)D_MODEL * F_OUT * 2);
    __bf16* xn_h    = (__bf16*)alloc((size_t)L_SEQ * D_MODEL * 2);
    float*  proj    = (float*) alloc((size_t)L_SEQ * F_IN * 4);
    __bf16* qh      = (__bf16*)alloc((size_t)NH * L_SEQ * HE * 2);
    __bf16* kh      = (__bf16*)alloc((size_t)NH * L_SEQ * HE * 2);
    __bf16* vh      = (__bf16*)alloc((size_t)NH * L_SEQ * HE * 2);
    __bf16* hbuf    = (__bf16*)alloc((size_t)L_SEQ * F_OUT * 2);

    // 1-2: weight conversion to bf16
    cvt_bf16_kernel<<<4096, 256, 0, stream>>>(w_in,  w_in_h,  F_IN * D_MODEL);
    cvt_bf16_kernel<<<4096, 256, 0, stream>>>(w_out, w_out_h, D_MODEL * F_OUT);

    // 3: RMSNorm + scale
    rmsnorm_kernel<<<L_SEQ, 256, 0, stream>>>(x, scale, xn_h);

    // 4: proj = xn @ w_in^T   (M=2048, N=7168, K=1024)
    gemm_bf16_kernel<<<dim3(L_SEQ / 128, F_IN / 64), 256, 0, stream>>>(
        xn_h, w_in_h, proj, nullptr, L_SEQ, F_IN, D_MODEL, 0);

    // 5: q/k norm + RoPE, v passthrough (head-major bf16)
    qkv_post_kernel<<<dim3(L_SEQ, NH), HE, 0, stream>>>(
        proj, theta, attn_scale, qh, kh, vh);

    // 6: GELU(up) -> h[:, D:5D]
    gelu_up_kernel<<<4096, 256, 0, stream>>>(proj, hbuf, L_SEQ * 4 * D_MODEL);

    // 7: causal flash attention -> h[:, 0:D]
    attn_kernel<<<dim3(L_SEQ / 16, NH), 32, 0, stream>>>(qh, kh, vh, hbuf);

    // 8: out = h @ w_out^T + x   (M=2048, N=1024, K=5120)
    gemm_bf16_kernel<<<dim3(L_SEQ / 128, D_MODEL / 64), 256, 0, stream>>>(
        hbuf, w_out_h, out, x, L_SEQ, D_MODEL, F_OUT, 1);
}
